// MPNN_12429635355003
// MI455X (gfx1250) — matrix operand order repaired and checked by reading, not compiled, not run on hardware
//
#include <hip/hip_runtime.h>

typedef __attribute__((ext_vector_type(16))) _Float16 v16h;
typedef __attribute__((ext_vector_type(8)))  _Float16 v8h;
typedef __attribute__((ext_vector_type(4)))  _Float16 v4h;
typedef __attribute__((ext_vector_type(8)))  float    v8f;
typedef __attribute__((ext_vector_type(4)))  float    v4f;

#define HIDDEN   64
#define MSG      64
#define EDGE_DIM 8
#define KMSG     136    // 2*HIDDEN + EDGE_DIM
#define KUPD     128    // MSG + HIDDEN
#define FS_MSG   144    // feature row stride (halfs): 72 DWORDs -> 2-way banks, 32B aligned rows
#define WS_MSG   176    // msg weight row stride (halfs): covers K=160 pad, 88 DWORDs -> 2-way
#define FS_UPD   144
#define WS_UPD   144

// ---------------------------------------------------------------------------
__global__ __launch_bounds__(256)
void mpnn_zero_f32(float* __restrict__ p, int n) {
    int i = blockIdx.x * 256 + threadIdx.x;
    if (i < n) p[i] = 0.0f;
}

__device__ __forceinline__ v4h cvt4(v4f v) {
    v4h o;
#pragma unroll
    for (int j = 0; j < 4; ++j) o[j] = (_Float16)v[j];
    return o;
}

// Permute 4-half group index so each lane's 16 A-fragment halves are contiguous:
// within every 32-half chunk (8 groups) swap blocks {2,3} <-> {4,5}
// (store order per chunk becomes k = [0-7, 16-23, 8-15, 24-31]).
__device__ __forceinline__ int permg(int g) {
    int b = g & 6;
    int pb = (b == 2) ? 4 : ((b == 4) ? 2 : b);
    return (g & ~6) | pb;
}

// ---------------------------------------------------------------------------
// Message kernel: m = relu(concat(h[src], h[dst], he) @ W + b), scatter-add to
// m_sum[dst] via f32 global atomics (m_sum = 25.6MB, resident in 192MB L2).
// Block = 256 threads = 8 waves; wave owns 16 edges (WMMA M).
// ---------------------------------------------------------------------------
__global__ __launch_bounds__(256)
void mpnn_msg_kernel(const float* __restrict__ h,     // [NN,64]
                     const float* __restrict__ he,    // [NE,8]
                     const float* __restrict__ W,     // [136,64] layer slice
                     const float* __restrict__ bias,  // [64]
                     const int*   __restrict__ src,
                     const int*   __restrict__ dst,
                     float*       __restrict__ msum,  // [NN,64] pre-zeroed
                     int NE)
{
    __shared__ __attribute__((aligned(32))) _Float16 sWT[MSG * WS_MSG]; // W^T [n][k<=160]
    __shared__ __attribute__((aligned(32))) _Float16 sF [128 * FS_MSG]; // features, perm layout
    __shared__ float sB[MSG];
    __shared__ int   sSrc[128], sDst[128];

    const int tid = threadIdx.x;
    const int e0  = blockIdx.x * 128;

    // Stage W transposed -> [n][k] f16 (permg k order, matches A); zero K pad rows 136..159.
    for (int idx = tid; idx < KMSG * MSG; idx += 256) {
        int k = idx >> 6, n = idx & 63;
        sWT[n * WS_MSG + ((permg(k >> 2) << 2) | (k & 3))] = (_Float16)W[idx];
    }
    for (int idx = tid; idx < 24 * MSG; idx += 256) {
        int k = KMSG + (idx >> 6), n = idx & 63;
        sWT[n * WS_MSG + k] = (_Float16)0.0f;
    }
    if (tid < MSG) sB[tid] = bias[tid];
    if (tid < 128) {
        int e = e0 + tid;
        sSrc[tid] = (e < NE) ? src[e] : 0;
        sDst[tid] = (e < NE) ? dst[e] : 0;
    }
    __syncthreads();

    // Stage features [128][144] f16 in A-fragment-contiguous (permuted) layout.
    for (int idx = tid; idx < 128 * 16; idx += 256) {            // h[src]: k 0..63
        int el = idx >> 4, g = idx & 15;
        v4f v = (e0 + el < NE) ? *(const v4f*)(h + (size_t)sSrc[el] * 64 + (g << 2)) : (v4f){};
        *(v4h*)&sF[el * FS_MSG + (permg(g) << 2)] = cvt4(v);
    }
    for (int idx = tid; idx < 128 * 16; idx += 256) {            // h[dst]: k 64..127
        int el = idx >> 4, g = 16 + (idx & 15);
        v4f v = (e0 + el < NE) ? *(const v4f*)(h + (size_t)sDst[el] * 64 + ((idx & 15) << 2)) : (v4f){};
        *(v4h*)&sF[el * FS_MSG + (permg(g) << 2)] = cvt4(v);
    }
    for (int idx = tid; idx < 128 * 2; idx += 256) {             // he: k 128..135 (perm = id)
        int el = idx >> 1, q = (idx & 1) << 2;
        v4f v = (e0 + el < NE) ? *(const v4f*)(he + (size_t)(e0 + el) * 8 + q) : (v4f){};
        *(v4h*)&sF[el * FS_MSG + 128 + q] = cvt4(v);
    }
    __syncthreads();

    const int lane = tid & 31;
    const int wv   = tid >> 5;
    const int l16  = lane & 15;   // A row / B-D column
    const int hi   = lane >> 4;   // half-wave selector

    // A fragments: one contiguous 32B v16h load per chunk (perm layout).
    const _Float16* fb = &sF[(wv * 16 + l16) * FS_MSG];
    v16h a[5];
#pragma unroll
    for (int kc = 0; kc < 4; ++kc)
        a[kc] = *(const v16h*)(fb + kc * 32 + hi * 16);
    // Chunk 4 (k=128..159): only k=128..135 nonzero (edge feats), held by hi==0 lanes.
    a[4] = (v16h){};
    if (hi == 0) {
        v8h lo = *(const v8h*)(fb + 128);
#pragma unroll
        for (int j = 0; j < 8; ++j) a[4][j] = lo[j];
    }

    // Accumulators seeded with bias (added exactly once across k accumulation).
    v8f acc[4];
#pragma unroll
    for (int nt = 0; nt < 4; ++nt) {
        float bv = sB[nt * 16 + l16];
#pragma unroll
        for (int r = 0; r < 8; ++r) acc[nt][r] = bv;
    }

    // Explicit double-buffered B fragments: issue chunk kc+1 loads before chunk kc WMMAs.
    const _Float16* wb = &sWT[l16 * WS_MSG + hi * 16];
    v16h bufA[4], bufB[4];
#pragma unroll
    for (int nt = 0; nt < 4; ++nt)
        bufA[nt] = *(const v16h*)(wb + nt * 16 * WS_MSG);
#pragma unroll
    for (int kc = 0; kc < 5; ++kc) {
        v16h (&cur)[4] = (kc & 1) ? bufB : bufA;
        v16h (&nxt)[4] = (kc & 1) ? bufA : bufB;
        if (kc + 1 < 5) {
#pragma unroll
            for (int nt = 0; nt < 4; ++nt)
                nxt[nt] = *(const v16h*)(wb + nt * 16 * WS_MSG + (kc + 1) * 32);
        }
#pragma unroll
        for (int nt = 0; nt < 4; ++nt)
            acc[nt] = __builtin_amdgcn_wmma_f32_16x16x32_f16(
                false, a[kc], false, cur[nt], (short)0, acc[nt], false, false);
    }

    // D[hi*8+r][l16]: relu, scatter-add to m_sum[dst].
#pragma unroll
    for (int nt = 0; nt < 4; ++nt) {
#pragma unroll
        for (int r = 0; r < 8; ++r) {
            int m = hi * 8 + r;
            int e = e0 + wv * 16 + m;
            if (e < NE) {
                float v = acc[nt][r];
                v = v > 0.0f ? v : 0.0f;
                atomicAdd(&msum[(size_t)sDst[wv * 16 + m] * 64 + nt * 16 + l16], v);
            }
        }
    }
}

// ---------------------------------------------------------------------------
// Node update: h' = relu(concat(m_sum, h) @ W_h + b_h), K = 128 exact.
// ---------------------------------------------------------------------------
__global__ __launch_bounds__(256)
void mpnn_upd_kernel(const float* __restrict__ msum,  // [NN,64]
                     const float* __restrict__ hin,   // [NN,64]
                     const float* __restrict__ W,     // [128,64]
                     const float* __restrict__ bias,  // [64]
                     float*       __restrict__ hout,  // [NN,64]
                     int NN)
{
    __shared__ __attribute__((aligned(32))) _Float16 sWT[HIDDEN * WS_UPD];
    __shared__ __attribute__((aligned(32))) _Float16 sF [128 * FS_UPD];
    __shared__ float sB[HIDDEN];

    const int tid = threadIdx.x;
    const int n0  = blockIdx.x * 128;

    for (int idx = tid; idx < KUPD * HIDDEN; idx += 256) {
        int k = idx >> 6, n = idx & 63;
        sWT[n * WS_UPD + ((permg(k >> 2) << 2) | (k & 3))] = (_Float16)W[idx];
    }
    if (tid < HIDDEN) sB[tid] = bias[tid];
    __syncthreads();

    for (int idx = tid; idx < 128 * 16; idx += 256) {            // m_sum: k 0..63
        int nl = idx >> 4, g = idx & 15;
        v4f v = (n0 + nl < NN) ? *(const v4f*)(msum + (size_t)(n0 + nl) * 64 + (g << 2)) : (v4f){};
        *(v4h*)&sF[nl * FS_UPD + (permg(g) << 2)] = cvt4(v);
    }
    for (int idx = tid; idx < 128 * 16; idx += 256) {            // h: k 64..127
        int nl = idx >> 4, g = 16 + (idx & 15);
        v4f v = (n0 + nl < NN) ? *(const v4f*)(hin + (size_t)(n0 + nl) * 64 + ((idx & 15) << 2)) : (v4f){};
        *(v4h*)&sF[nl * FS_UPD + (permg(g) << 2)] = cvt4(v);
    }
    __syncthreads();

    const int lane = tid & 31;
    const int wv   = tid >> 5;
    const int l16  = lane & 15;
    const int hi   = lane >> 4;

    const _Float16* fb = &sF[(wv * 16 + l16) * FS_UPD];
    v16h a[4];
#pragma unroll
    for (int kc = 0; kc < 4; ++kc)
        a[kc] = *(const v16h*)(fb + kc * 32 + hi * 16);

    v8f acc[4];
#pragma unroll
    for (int nt = 0; nt < 4; ++nt) {
        float bv = sB[nt * 16 + l16];
#pragma unroll
        for (int r = 0; r < 8; ++r) acc[nt][r] = bv;
    }

    const _Float16* wb = &sWT[l16 * WS_UPD + hi * 16];
    v16h bufA[4], bufB[4];
#pragma unroll
    for (int nt = 0; nt < 4; ++nt)
        bufA[nt] = *(const v16h*)(wb + nt * 16 * WS_UPD);
#pragma unroll
    for (int kc = 0; kc < 4; ++kc) {
        v16h (&cur)[4] = (kc & 1) ? bufB : bufA;
        v16h (&nxt)[4] = (kc & 1) ? bufA : bufB;
        if (kc + 1 < 4) {
#pragma unroll
            for (int nt = 0; nt < 4; ++nt)
                nxt[nt] = *(const v16h*)(wb + nt * 16 * WS_UPD + (kc + 1) * 32);
        }
#pragma unroll
        for (int nt = 0; nt < 4; ++nt)
            acc[nt] = __builtin_amdgcn_wmma_f32_16x16x32_f16(
                false, a[kc], false, cur[nt], (short)0, acc[nt], false, false);
    }

#pragma unroll
    for (int nt = 0; nt < 4; ++nt) {
#pragma unroll
        for (int r = 0; r < 8; ++r) {
            int node = n0 + wv * 16 + hi * 8 + r;
            if (node < NN) {
                float v = acc[nt][r];
                hout[(size_t)node * 64 + nt * 16 + l16] = v > 0.0f ? v : 0.0f;
            }
        }
    }
}

// ---------------------------------------------------------------------------
extern "C" void kernel_launch(void* const* d_in, const int* in_sizes, int n_in,
                              void* d_out, int out_size, void* d_ws, size_t ws_size,
                              hipStream_t stream)
{
    const float* h_n = (const float*)d_in[0];
    const float* h_e = (const float*)d_in[1];
    const float* W_m = (const float*)d_in[2];   // [3][136][64]
    const float* b_m = (const float*)d_in[3];   // [3][64]
    const float* W_h = (const float*)d_in[4];   // [3][128][64]
    const float* b_h = (const float*)d_in[5];   // [3][64]
    const int*   src = (const int*)d_in[6];
    const int*   dst = (const int*)d_in[7];
    float*       out = (float*)d_out;

    const int NN = in_sizes[0] / HIDDEN;   // 100000
    const int NE = in_sizes[6];            // 3200000

    float* msum = (float*)d_ws;                   // NN*64 f32, L2-resident
    float* hA   = msum + (size_t)NN * HIDDEN;     // ping
    float* hB   = hA   + (size_t)NN * HIDDEN;     // pong

    const int msgBlocks  = (NE + 127) / 128;
    const int updBlocks  = (NN + 127) / 128;
    const int zeroBlocks = (NN * HIDDEN + 255) / 256;

    const float* hcur = h_n;
    for (int l = 0; l < 3; ++l) {
        mpnn_zero_f32<<<zeroBlocks, 256, 0, stream>>>(msum, NN * HIDDEN);
        mpnn_msg_kernel<<<msgBlocks, 256, 0, stream>>>(
            hcur, h_e, W_m + (size_t)l * KMSG * MSG, b_m + (size_t)l * MSG,
            src, dst, msum, NE);
        float* hnext = (l == 2) ? out : ((l == 0) ? hA : hB);
        mpnn_upd_kernel<<<updBlocks, 256, 0, stream>>>(
            msum, hcur, W_h + (size_t)l * KUPD * HIDDEN, b_h + (size_t)l * HIDDEN,
            hnext, NN);
        hcur = hnext;
    }
}
